// _NonLocalNd_18554258719141
// MI455X (gfx1250) — matrix operand order repaired and checked by reading, not compiled, not run on hardware
//
#include <hip/hip_runtime.h>

typedef __attribute__((ext_vector_type(16))) _Float16 v16h;
typedef __attribute__((ext_vector_type(2)))  __fp16   v2h;   // cvt_pkrtz result type
typedef __attribute__((ext_vector_type(8)))  float    v8f;

static constexpr int HEADS = 2;

union U4H8 { uint4 u; _Float16 h[8]; };
union H8U4 { v2h p[4]; uint4 u; };

__device__ __forceinline__ v16h frag16(const _Float16* p0, const _Float16* p1) {
  U4H8 a, b;
  a.u = *(const uint4*)p0;
  b.u = *(const uint4*)p1;
  v16h r;
#pragma unroll
  for (int i = 0; i < 8; ++i) { r[i] = a.h[i]; r[8 + i] = b.h[i]; }
  return r;
}

__device__ __forceinline__ v8f wmma_f16(v16h a, v16h b, v8f c) {
  return __builtin_amdgcn_wmma_f32_16x16x32_f16(
      false, a, false, b, (short)0, c, false, false);
}

// gfx1250 async global->LDS copy (ASYNCcnt), no VGPR round-trip.
__device__ __forceinline__ void async_to_lds_b128(void* lds_ptr, const void* gptr) {
  uint32_t l = (uint32_t)(uintptr_t)lds_ptr;  // low 32 bits of generic ptr = LDS offset
  asm volatile("global_load_async_to_lds_b128 %0, %1, off"
               :: "v"(l), "v"(gptr) : "memory");
}
__device__ __forceinline__ void wait_async0() {
  asm volatile("s_wait_asynccnt 0" ::: "memory");
}

// ---------------------------------------------------------------------------
// One-shot f32 -> f16 conversion (makes hot GEMM operands f16 + L2 resident)
// ---------------------------------------------------------------------------
__global__ void __launch_bounds__(256)
cvt_f32_to_f16_kernel(const float* __restrict__ src, _Float16* __restrict__ dst,
                      long n8) {
  long i = (long)blockIdx.x * 256 + threadIdx.x;
  if (i >= n8) return;
  const float4 a = ((const float4*)src)[i * 2];
  const float4 b = ((const float4*)src)[i * 2 + 1];
  H8U4 o;
  o.p[0] = __builtin_amdgcn_cvt_pkrtz(a.x, a.y);
  o.p[1] = __builtin_amdgcn_cvt_pkrtz(a.z, a.w);
  o.p[2] = __builtin_amdgcn_cvt_pkrtz(b.x, b.y);
  o.p[3] = __builtin_amdgcn_cvt_pkrtz(b.z, b.w);
  ((uint4*)dst)[i] = o.u;
}

// ---------------------------------------------------------------------------
// Tiled WMMA GEMM, double-buffered LDS, async staging where operands are f16.
//   A: [M,K] row-major; f16 (A_F16, async) or f32 (converted while staging)
//   B: f16. B_KC: stored [n,K] (K contig, async). else [K,n] (transpose-staged)
//   BIAS_MODE: 0 = bias[col], 1 = bias[row]
//   OUT_MODE: 0 f16 [M,N]; 1 f32 [M,N]; 2 f16 head-split [(z*H+row/dh), col, row%dh]
// ---------------------------------------------------------------------------
template <int OUT_MODE, int BIAS_MODE, bool B_KC, bool A_F16, bool HAS_RES>
__global__ void __launch_bounds__(256)
gemm_wmma_kernel(const void* __restrict__ Av, long lda, long a_bs,
                 const _Float16* __restrict__ B, long ldb, long b_bs,
                 const float* __restrict__ bias,
                 const float* __restrict__ res, long res_bs,
                 void* __restrict__ Out, long out_bs,
                 int Ncols, int K, int dhead) {
  constexpr int BK = 32, LDK = 40;
  __shared__ _Float16 As[2][128][LDK];
  __shared__ _Float16 Bs[2][128][LDK];

  const int t    = threadIdx.x;
  const int wave = t >> 5;
  const int lane = t & 31;
  const int lr   = lane & 15;
  const int lh   = lane >> 4;
  const int wm   = wave >> 1;
  const int wn   = wave & 1;
  const int m0   = blockIdx.y * 128;
  const int n0   = blockIdx.x * 128;
  const int z    = blockIdx.z;

  const _Float16* Bb = B + (long)z * b_bs;

  auto stage = [&](int buf, int k0) {
    if constexpr (A_F16) {
      const _Float16* Ah = (const _Float16*)Av + (long)z * a_bs;
#pragma unroll
      for (int it = 0; it < 2; ++it) {
        int li = (it * 256 + t) * 8;
        int mr = li >> 5, kk = li & 31;
        async_to_lds_b128(&As[buf][mr][kk], Ah + (long)(m0 + mr) * lda + k0 + kk);
      }
    } else {
      const float* Af = (const float*)Av + (long)z * a_bs;
#pragma unroll
      for (int it = 0; it < 4; ++it) {
        int li = (it * 256 + t) * 4;
        int mr = li >> 5, kk = li & 31;
        const float4 v = *(const float4*)(Af + (long)(m0 + mr) * lda + k0 + kk);
        *(v2h*)&As[buf][mr][kk]     = __builtin_amdgcn_cvt_pkrtz(v.x, v.y);
        *(v2h*)&As[buf][mr][kk + 2] = __builtin_amdgcn_cvt_pkrtz(v.z, v.w);
      }
    }
    if constexpr (B_KC) {
#pragma unroll
      for (int it = 0; it < 2; ++it) {
        int li = (it * 256 + t) * 8;
        int nr = li >> 5, kk = li & 31;
        async_to_lds_b128(&Bs[buf][nr][kk], Bb + (long)(n0 + nr) * ldb + k0 + kk);
      }
    } else {  // B stored [K,N], N contiguous: transpose while staging
#pragma unroll
      for (int it = 0; it < 2; ++it) {
        int li = (it * 256 + t) * 8;
        int kr = li >> 7, nc = li & 127;
        U4H8 v;
        v.u = *(const uint4*)(Bb + (long)(k0 + kr) * ldb + n0 + nc);
#pragma unroll
        for (int x = 0; x < 8; ++x) Bs[buf][nc + x][kr] = v.h[x];
      }
    }
  };

  v8f acc[2][4];
#pragma unroll
  for (int i = 0; i < 2; ++i)
#pragma unroll
    for (int j = 0; j < 4; ++j) acc[i][j] = {};

  stage(0, 0);
  const int nk = K / BK;
  for (int kb = 0; kb < nk; ++kb) {
    const int cur = kb & 1;
    wait_async0();
    __syncthreads();
    if (kb + 1 < nk) stage(cur ^ 1, (kb + 1) * BK);  // prefetch next tile

    v16h af[2], bfr[4];
#pragma unroll
    for (int tm = 0; tm < 2; ++tm) {
      const _Float16* p = &As[cur][wm * 32 + tm * 16 + lr][lh * 8];
      af[tm] = frag16(p, p + 16);
    }
#pragma unroll
    for (int tn = 0; tn < 4; ++tn) {
      const _Float16* p = &Bs[cur][wn * 64 + tn * 16 + lr][lh * 8];
      bfr[tn] = frag16(p, p + 16);
    }
#pragma unroll
    for (int tm = 0; tm < 2; ++tm)
#pragma unroll
      for (int tn = 0; tn < 4; ++tn)
        acc[tm][tn] = wmma_f16(af[tm], bfr[tn], acc[tm][tn]);
  }

#pragma unroll
  for (int tm = 0; tm < 2; ++tm)
#pragma unroll
    for (int tn = 0; tn < 4; ++tn)
#pragma unroll
      for (int r = 0; r < 8; ++r) {
        int row = m0 + wm * 32 + tm * 16 + r + 8 * lh;
        int col = n0 + wn * 64 + tn * 16 + lr;
        float v = acc[tm][tn][r];
        if constexpr (BIAS_MODE == 0) v += bias[col];
        else                          v += bias[row];
        if constexpr (HAS_RES)
          v += res[(long)z * res_bs + (long)row * Ncols + col];
        if constexpr (OUT_MODE == 0) {
          ((_Float16*)Out)[(long)z * out_bs + (long)row * Ncols + col] = (_Float16)v;
        } else if constexpr (OUT_MODE == 1) {
          ((float*)Out)[(long)z * out_bs + (long)row * Ncols + col] = v;
        } else {
          int hh = row / dhead, dd = row % dhead;
          long idx = (((long)(z * HEADS + hh) * Ncols) + col) * dhead + dd;
          ((_Float16*)Out)[idx] = (_Float16)v;
        }
      }
}

// ---------------------------------------------------------------------------
// Flash attention: (bh, 64-row query block) x streamed 32-key blocks.
//   Th/Ph/G: [BH, N, d] f16 (d contiguous).  Yt: [B, N, HEADS*d] f16.
// ---------------------------------------------------------------------------
__global__ void __launch_bounds__(256)
flash_attn_kernel(const _Float16* __restrict__ Th,
                  const _Float16* __restrict__ Ph,
                  const _Float16* __restrict__ G,
                  _Float16* __restrict__ Yt,
                  int Nsp, float scale) {
  constexpr int QB = 64, KB = 32, D = 128, DP = D + 8, KP = KB + 8;
  __shared__ _Float16 Qs[QB][DP];
  __shared__ _Float16 Ks[KB][DP];
  __shared__ _Float16 Gts[D][KP];
  __shared__ float    Ss[QB][KB + 4];
  __shared__ _Float16 Ps[QB][KP];
  __shared__ float    arow[QB];
  __shared__ float    lrow[QB];

  const int t    = threadIdx.x;
  const int wave = t >> 5;
  const int lane = t & 31;
  const int lr   = lane & 15;
  const int lh   = lane >> 4;
  const int bh   = blockIdx.y;
  const int q0   = blockIdx.x * QB;
  const long base = (long)bh * Nsp * D;

  // async-load Q block (64 x 128); completion covered by first wait_async0
#pragma unroll
  for (int it = 0; it < 4; ++it) {
    int li = (it * 256 + t) * 8;
    int qr = li >> 7, dd = li & 127;
    async_to_lds_b128(&Qs[qr][dd], Th + base + (long)(q0 + qr) * D + dd);
  }

  const int sm = wave >> 1;
  const int sn = wave & 1;

  v8f yacc[4];
#pragma unroll
  for (int j = 0; j < 4; ++j) yacc[j] = {};
  float m_i = -3.0e38f, l_i = 0.f;  // live in threads t < 64 (row = t)

  for (int j0 = 0; j0 < Nsp; j0 += KB) {
    __syncthreads();  // previous block's LDS readers are done
    // async-stage K block (32 x 128)
#pragma unroll
    for (int it = 0; it < 2; ++it) {
      int li = (it * 256 + t) * 8;
      int kr = li >> 7, dd = li & 127;
      async_to_lds_b128(&Ks[kr][dd], Ph + base + (long)(j0 + kr) * D + dd);
    }
    // stage G block transposed: Gts[d][key] (element scatter; stays sync)
#pragma unroll
    for (int it = 0; it < 2; ++it) {
      int li = (it * 256 + t) * 8;
      int kr = li >> 7, dd = li & 127;
      U4H8 v;
      v.u = *(const uint4*)(G + base + (long)(j0 + kr) * D + dd);
#pragma unroll
      for (int x = 0; x < 8; ++x) Gts[dd + x][kr] = v.h[x];
    }
    wait_async0();
    __syncthreads();
    // S tile = Q * K^T  (16x16, K=128 -> 4 WMMAs per wave)
    v8f s = {};
#pragma unroll
    for (int kk = 0; kk < D; kk += 32) {
      const _Float16* pq = &Qs[sm * 16 + lr][kk + lh * 8];
      const _Float16* pk = &Ks[sn * 16 + lr][kk + lh * 8];
      s = wmma_f16(frag16(pq, pq + 16), frag16(pk, pk + 16), s);
    }
#pragma unroll
    for (int r = 0; r < 8; ++r)
      Ss[sm * 16 + r + 8 * lh][sn * 16 + lr] = s[r];
    __syncthreads();
    // online softmax over this 32-key block (one thread per query row)
    if (t < QB) {
      int r = t;
      float mx = m_i;
#pragma unroll
      for (int c = 0; c < KB; ++c) mx = fmaxf(mx, Ss[r][c] * scale);
      float alpha = __expf(m_i - mx);
      float sum = 0.f;
#pragma unroll
      for (int c = 0; c < KB; ++c) {
        float p = __expf(Ss[r][c] * scale - mx);
        sum += p;
        Ps[r][c] = (_Float16)p;
      }
      l_i = l_i * alpha + sum;
      m_i = mx;
      arow[r] = alpha;
    }
    __syncthreads();
    // rescale running output, then Y += P * G
    const _Float16* pp = &Ps[sm * 16 + lr][lh * 8];
    v16h ap = frag16(pp, pp + 16);
#pragma unroll
    for (int j = 0; j < 4; ++j) {
#pragma unroll
      for (int r = 0; r < 8; ++r)
        yacc[j][r] *= arow[sm * 16 + r + 8 * lh];
      const _Float16* pg = &Gts[sn * 64 + j * 16 + lr][lh * 8];
      yacc[j] = wmma_f16(ap, frag16(pg, pg + 16), yacc[j]);
    }
  }
  __syncthreads();
  if (t < QB) lrow[t] = l_i;
  __syncthreads();

  const int b = bh >> 1, hh = bh & 1;
#pragma unroll
  for (int j = 0; j < 4; ++j)
#pragma unroll
    for (int r = 0; r < 8; ++r) {
      int row = sm * 16 + r + 8 * lh;
      int dd  = sn * 64 + j * 16 + lr;
      float v = yacc[j][r] / lrow[row];
      Yt[((long)b * Nsp + q0 + row) * (HEADS * D) + hh * D + dd] = (_Float16)v;
    }
}

// ---------------------------------------------------------------------------
extern "C" void kernel_launch(void* const* d_in, const int* in_sizes, int n_in,
                              void* d_out, int out_size, void* d_ws, size_t ws_size,
                              hipStream_t stream) {
  (void)in_sizes; (void)n_in; (void)out_size; (void)ws_size;
  const float* x     = (const float*)d_in[0];
  const float* oa    = (const float*)d_in[1];
  const float* mlp_w = (const float*)d_in[2];
  const float* mlp_b = (const float*)d_in[3];
  const float* g_w   = (const float*)d_in[4];
  const float* g_b   = (const float*)d_in[5];
  const float* th_w  = (const float*)d_in[6];
  const float* th_b  = (const float*)d_in[7];
  const float* ph_w  = (const float*)d_in[8];
  const float* ph_b  = (const float*)d_in[9];
  const float* o_w   = (const float*)d_in[10];
  const float* o_b   = (const float*)d_in[11];
  float* out = (float*)d_out;

  const int B = 4, C = 512, N = 4096, IC = 256, dh = 128;
  const long XN = (long)B * C * N;       // 8.4M
  const long WN = (long)N * N;           // 16.8M

  char* ws = (char*)d_ws;
  _Float16* xf  = (_Float16*)ws; ws += XN * 2;                          // [B,C,N]
  _Float16* xh  = (_Float16*)ws; ws += XN * 2;                          // x (f16)
  _Float16* oah = (_Float16*)ws; ws += XN * 2;                          // oa (f16)
  _Float16* mwh = (_Float16*)ws; ws += WN * 2;                          // mlp_w (f16)
  _Float16* tht = (_Float16*)ws; ws += (size_t)B * HEADS * N * dh * 2;  // [BH,N,d]
  _Float16* pht = (_Float16*)ws; ws += (size_t)B * HEADS * N * dh * 2;
  _Float16* gt  = (_Float16*)ws; ws += (size_t)B * HEADS * N * dh * 2;
  _Float16* yt  = (_Float16*)ws; ws += (size_t)B * N * IC * 2;          // [B,N,IC]

  // 0) one-shot f32->f16 conversions (hot operands become L2-resident f16)
  cvt_f32_to_f16_kernel<<<(int)(XN / 8 / 256), 256, 0, stream>>>(x, xh, XN / 8);
  cvt_f32_to_f16_kernel<<<(int)(XN / 8 / 256), 256, 0, stream>>>(oa, oah, XN / 8);
  cvt_f32_to_f16_kernel<<<(int)(WN / 8 / 256), 256, 0, stream>>>(mlp_w, mwh, WN / 8);

  // 1) remix: xf = x + x @ mlp_w^T + mlp_b (A,B f16 async-staged)
  gemm_wmma_kernel<0, 0, true, true, true>
      <<<dim3(N / 128, (B * C) / 128, 1), 256, 0, stream>>>(
          xh, N, 0, mwh, N, 0, mlp_b, x, 0, xf, 0, N, N, dh);

  // 2) theta conv on output_attn (f16) -> [BH,N,d]
  gemm_wmma_kernel<2, 1, false, false, false>
      <<<dim3(N / 128, IC / 128, B), 256, 0, stream>>>(
          th_w, C, 0, oah, N, (long)C * N, th_b, nullptr, 0, tht, 0, N, C, dh);

  // 3) g conv on xf -> [BH,N,d]
  gemm_wmma_kernel<2, 1, false, false, false>
      <<<dim3(N / 128, IC / 128, B), 256, 0, stream>>>(
          g_w, C, 0, xf, N, (long)C * N, g_b, nullptr, 0, gt, 0, N, C, dh);

  // 4) phi conv on xf -> [BH,N,d]
  gemm_wmma_kernel<2, 1, false, false, false>
      <<<dim3(N / 128, IC / 128, B), 256, 0, stream>>>(
          ph_w, C, 0, xf, N, (long)C * N, ph_b, nullptr, 0, pht, 0, N, C, dh);

  // 5) flash attention -> yt [B,N,IC]
  flash_attn_kernel<<<dim3(N / 64, B * HEADS), 256, 0, stream>>>(
      tht, pht, gt, yt, N, 0.08838834764831845f);

  // 6) out conv + residual -> f32 output
  gemm_wmma_kernel<1, 1, true, false, true>
      <<<dim3(N / 128, C / 128, B), 256, 0, stream>>>(
          o_w, IC, 0, yt, IC, (long)N * IC, o_b, oa, (long)C * N,
          out, (long)C * N, N, IC, dh);
}
